// AMGCN_seg_EdgeConv_146028888442
// MI455X (gfx1250) — matrix operand order repaired and checked by reading, not compiled, not run on hardware
//
#include <hip/hip_runtime.h>
#include <cfloat>
#include <cmath>

typedef __attribute__((ext_vector_type(16))) _Float16 v16h;
typedef __attribute__((ext_vector_type(8)))  _Float16 v8h;
typedef __attribute__((ext_vector_type(8)))  float    v8f;

union Frag { v16h v; v8h h8[2]; };

#define NPTS 4096

// ------------------------------------------- f32 -> f16 with zero padding (rows/cols)
__global__ void cvt_pad_kernel(const float* __restrict__ src, _Float16* __restrict__ dst,
                               int srcRows, int srcCols, int dstRows, int dstCols)
{
    int t = blockIdx.x * blockDim.x + threadIdx.x;
    if (t >= dstRows * dstCols) return;
    int r = t / dstCols, c = t % dstCols;
    float v = (r < srcRows && c < srcCols) ? src[(size_t)r * srcCols + c] : 0.0f;
    dst[t] = (_Float16)v;
}

// ------------------------------------------- WMMA GEMM: C = A @ W^T (+bias, bn, act)
// A: [M,K] f16, M % 16 == 0, K % 32 == 0 (zero-padded).  W: [Npad,K] f16, Npad % 16 == 0.
// C: [M,Nout] f32 (Nout <= Npad; store-guarded). One wave per 16x16 tile, no EXEC
// manipulation in the K loop: fragments are built from contiguous 16-byte loads
// (CDNA5 ISA 7.12.2 layouts are piecewise-contiguous in K).
__global__ __launch_bounds__(32)
void gemm_wmma_kernel(const _Float16* __restrict__ A, const _Float16* __restrict__ W,
                      const float* __restrict__ bias, const float* __restrict__ bng,
                      const float* __restrict__ bnb, float* __restrict__ C,
                      int K, int Nout, int act)
{
    const int tn = blockIdx.x, tm = blockIdx.y;
    const int lane = threadIdx.x & 31;
    const int grp  = lane >> 4;      // half-wave group
    const int l16  = lane & 15;
    const int arow = tm * 16 + l16;  // A row held by this lane
    const int wrow = tn * 16 + l16;  // W row (= output column) held by this lane
    const v8h* __restrict__ ap = (const v8h*)(A + (size_t)arow * K);
    const v8h* __restrict__ wp = (const v8h*)(W + (size_t)wrow * K);
    v8f acc = {};
    const int nk = K >> 5;           // number of 32-wide K steps
#pragma unroll 2
    for (int kb = 0; kb < nk; ++kb) {
        Frag a, b;
        // A 16-bit 16x32 fragment: halves [8*grp, +8) and [16+8*grp, +8)
        a.h8[0] = ap[kb * 4 + grp];
        a.h8[1] = ap[kb * 4 + grp + 2];
        // B 32x16 fragment: halves [8*grp, +8) and [16+8*grp, +8)
        b.h8[0] = wp[kb * 4 + grp];
        b.h8[1] = wp[kb * 4 + grp + 2];
        acc = __builtin_amdgcn_wmma_f32_16x16x32_f16(false, a.v, false, b.v,
                                                     (short)0, acc, false, false);
    }
    const int cc = tn * 16 + l16;
    if (cc < Nout) {
        const float bi = bias ? bias[cc] : 0.0f;
        const float g  = bng  ? bng[cc]  : 1.0f;
        const float bb = bng  ? bnb[cc]  : 0.0f;
#pragma unroll
        for (int v = 0; v < 8; ++v) {
            int rr = tm * 16 + grp * 8 + v;   // C/D layout: VGPR v <-> row v + 8*grp
            float val = (acc[v] + bi) * g + bb;
            if (act == 1) val = fmaxf(val, 0.0f);
            C[(size_t)rr * Nout + cc] = val;
        }
    }
}

// ------------------------------------------------- small f32 linear: Y = X @ W^T + b
__global__ void linear_small_kernel(const float* __restrict__ X, const float* __restrict__ W,
                                    const float* __restrict__ bias, float* __restrict__ Y,
                                    int M, int N, int K, int act)
{
    int t = blockIdx.x * blockDim.x + threadIdx.x;
    if (t >= M * N) return;
    int r = t / N, c = t % N;
    const float* x = X + (size_t)r * K;
    const float* w = W + (size_t)c * K;
    float s = bias ? bias[c] : 0.0f;
    for (int k = 0; k < K; ++k) s += x[k] * w[k];
    if (act == 1) s = fmaxf(s, 0.0f);
    Y[(size_t)r * N + c] = s;
}

// ------------------------------------------------- kNN: one block per query row
__global__ void knn_kernel(const float* __restrict__ F, int N, int D, int k, int* __restrict__ out)
{
    __shared__ float sd[NPTS];
    __shared__ float rv[256];
    __shared__ int   ri[256];
    const int i = blockIdx.x;
    const float* fi = F + (size_t)i * D;
    for (int j = threadIdx.x; j < N; j += blockDim.x) {
        const float* fj = F + (size_t)j * D;
        float s = 0.0f;
        for (int c = 0; c < D; ++c) { float d = fj[c] - fi[c]; s += d * d; }
        sd[j] = s;
    }
    __syncthreads();
    for (int t = 0; t < k; ++t) {
        float bv = FLT_MAX; int bi = N;
        for (int j = threadIdx.x; j < N; j += blockDim.x) {
            float v = sd[j];
            if (v < bv || (v == bv && j < bi)) { bv = v; bi = j; }
        }
        rv[threadIdx.x] = bv; ri[threadIdx.x] = bi;
        __syncthreads();
        for (int s = 128; s > 0; s >>= 1) {
            if (threadIdx.x < (unsigned)s) {
                float v2 = rv[threadIdx.x + s]; int i2 = ri[threadIdx.x + s];
                if (v2 < rv[threadIdx.x] || (v2 == rv[threadIdx.x] && i2 < ri[threadIdx.x])) {
                    rv[threadIdx.x] = v2; ri[threadIdx.x] = i2;
                }
            }
            __syncthreads();
        }
        if (threadIdx.x == 0) { out[(size_t)i * k + t] = ri[0]; sd[ri[0]] = FLT_MAX; }
        __syncthreads();
    }
}

// ------------------------------------------------- EdgeConv epilogue:
// out_i[c] = max_j t[idx_ij][c] - t[i][c] + btheta[c] + p[i][c]
__global__ void edge_max_kernel(const float* __restrict__ t, const float* __restrict__ p,
                                const float* __restrict__ bth, const int* __restrict__ idx,
                                float* __restrict__ out, int N, int D, int k)
{
    int i = blockIdx.x, c = threadIdx.x;
    if (c >= D) return;
    const int* id = idx + (size_t)i * k;
    float m = -FLT_MAX;
    for (int j = 0; j < k; ++j) m = fmaxf(m, t[(size_t)id[j] * D + c]);
    out[(size_t)i * D + c] = m - t[(size_t)i * D + c] + bth[c] + p[(size_t)i * D + c];
}

// ------------------------------------------------- column-wise global max-pool
__global__ void rowmax_kernel(const float* __restrict__ X, float* __restrict__ out, int M, int C)
{
    int c = blockIdx.x * blockDim.x + threadIdx.x;
    if (c >= C) return;
    float m = -FLT_MAX;
    for (int r = 0; r < M; ++r) m = fmaxf(m, X[(size_t)r * C + c]);
    out[c] = m;
}

// ------------------------------------------------- GATv2: one block per (node, head), D=256
__global__ void gatv2_kernel(const float* __restrict__ src, const float* __restrict__ dst,
                             const int* __restrict__ idx, const float* __restrict__ attn,
                             const float* __restrict__ bias, float* __restrict__ out,
                             int N, int k)
{
    __shared__ float red[256];
    __shared__ float lg[128];
    const int i = blockIdx.x, h = blockIdx.y, tid = threadIdx.x;
    const int* id = idx + (size_t)i * k;
    const float dv = dst[(size_t)i * 1024 + h * 256 + tid];
    const float av = attn[h * 256 + tid];
    for (int j = 0; j < k; ++j) {
        float e = src[(size_t)id[j] * 1024 + h * 256 + tid] + dv;
        e = (e > 0.0f) ? e : 0.2f * e;      // leaky_relu(0.2)
        red[tid] = e * av;
        __syncthreads();
        for (int s = 128; s > 0; s >>= 1) {
            if (tid < (unsigned)s) red[tid] += red[tid + s];
            __syncthreads();
        }
        if (tid == 0) lg[j] = red[0];
        __syncthreads();
    }
    if (tid == 0) {
        float m = -FLT_MAX;
        for (int j = 0; j < k; ++j) m = fmaxf(m, lg[j]);
        float s = 0.0f;
        for (int j = 0; j < k; ++j) { lg[j] = expf(lg[j] - m); s += lg[j]; }
        float inv = 1.0f / s;
        for (int j = 0; j < k; ++j) lg[j] *= inv;
    }
    __syncthreads();
    float acc = 0.0f;
    for (int j = 0; j < k; ++j) acc += lg[j] * src[(size_t)id[j] * 1024 + h * 256 + tid];
    out[(size_t)i * 1024 + h * 256 + tid] = acc + bias[h * 256 + tid];
}

// ------------------------------------------------- 3x3 Householder QR (LAPACK sign convention)
__global__ void qr3_kernel(const float* __restrict__ m, float* __restrict__ q, float* __restrict__ r)
{
    if (threadIdx.x || blockIdx.x) return;
    float R[3][3], Q[3][3];
    for (int i = 0; i < 3; ++i)
        for (int j = 0; j < 3; ++j) { R[i][j] = m[i * 3 + j]; Q[i][j] = (i == j) ? 1.f : 0.f; }
    for (int c = 0; c < 2; ++c) {
        float nrm = 0.f;
        for (int i = c; i < 3; ++i) nrm += R[i][c] * R[i][c];
        nrm = sqrtf(nrm);
        if (nrm < 1e-30f) continue;
        float alpha = (R[c][c] >= 0.f) ? -nrm : nrm;
        float v[3] = {0.f, 0.f, 0.f};
        for (int i = c; i < 3; ++i) v[i] = R[i][c];
        v[c] -= alpha;
        float vn2 = 0.f;
        for (int i = c; i < 3; ++i) vn2 += v[i] * v[i];
        if (vn2 < 1e-30f) continue;
        float s = 2.f / vn2;
        for (int col = c; col < 3; ++col) {
            float d = 0.f;
            for (int i = c; i < 3; ++i) d += v[i] * R[i][col];
            d *= s;
            for (int i = c; i < 3; ++i) R[i][col] -= d * v[i];
        }
        for (int row = 0; row < 3; ++row) {
            float d = 0.f;
            for (int i = c; i < 3; ++i) d += Q[row][i] * v[i];
            d *= s;
            for (int i = c; i < 3; ++i) Q[row][i] -= d * v[i];
        }
    }
    for (int i = 0; i < 3; ++i)
        for (int j = 0; j < 3; ++j) { q[i*3+j] = Q[i][j]; r[i*3+j] = (j >= i) ? R[i][j] : 0.f; }
}

// ------------------------------------------------- x = pc @ q  (q row-major 3x3)
__global__ void rotate_kernel(const float* __restrict__ pc, const float* __restrict__ q,
                              float* __restrict__ x, int N)
{
    int i = blockIdx.x * blockDim.x + threadIdx.x;
    if (i >= N) return;
    float p0 = pc[i*3], p1 = pc[i*3+1], p2 = pc[i*3+2];
    for (int c = 0; c < 3; ++c)
        x[i*3+c] = p0 * q[0*3+c] + p1 * q[1*3+c] + p2 * q[2*3+c];
}

__global__ void extract2_kernel(const float* __restrict__ src, int c0, int c1,
                                float* __restrict__ dst, int N)
{
    int i = blockIdx.x * blockDim.x + threadIdx.x;
    if (i >= N) return;
    dst[i*2]   = src[i*3 + c0];
    dst[i*2+1] = src[i*3 + c1];
}

__global__ void copy_cols_kernel(const float* __restrict__ src, int srcStride,
                                 float* __restrict__ dst, int dstStride, int dstOff,
                                 int rows, int cols)
{
    int t = blockIdx.x * blockDim.x + threadIdx.x;
    if (t >= rows * cols) return;
    int r = t / cols, c = t % cols;
    dst[(size_t)r * dstStride + dstOff + c] = src[(size_t)r * srcStride + c];
}

__global__ void broadcast_row_kernel(const float* __restrict__ vec, float* __restrict__ dst,
                                     int dstStride, int dstOff, int rows, int len)
{
    int t = blockIdx.x * blockDim.x + threadIdx.x;
    if (t >= rows * len) return;
    int r = t / len, c = t % len;
    dst[(size_t)r * dstStride + dstOff + c] = vec[c];
}

__global__ void write_r_kernel(const float* __restrict__ r, float* __restrict__ out)
{
    int i = threadIdx.x;
    if (i < 9) out[i] = r[i];
}

// =========================================================================================
extern "C" void kernel_launch(void* const* d_in, const int* in_sizes, int n_in,
                              void* d_out, int out_size, void* d_ws, size_t ws_size,
                              hipStream_t stream)
{
    (void)in_sizes; (void)n_in; (void)out_size; (void)ws_size;
    const int N = NPTS;
    auto P = [&](int i) -> const float* { return (const float*)d_in[i]; };
    // d_in flattening (setup_inputs insertion order, w before b, theta before phi):
    // 0 pointcloud | tnet: 1 fc1.w 2 fc1.b 3 fc2.w 4 fc2.b 5 fc3.w 6 fc3.b 7 fc4.w 8 fc4.b
    // Gconv1: 9..12  Gconv2: 13..16  Gconv3: 17..20  Gconv4: 21..24  Gconv5: 25..28
    // Econv1xy: 29..32 Econv2xy: 33..36 Econv1yz: 37..40 Econv2yz: 41..44 Econv1xz: 45..48 Econv2xz: 49..52
    // fc0: 53 w 54 b | gat: 55 ws 56 wd 57 attn 58 bias | fc1: 59 w 60 b | bn1: 61 g 62 b
    // fc2: 63 w 64 b | bn2: 65 g 66 b | fc3: 67 w 68 b

    uint8_t* ws = (uint8_t*)d_ws;
    size_t off = 0;
    auto alloc = [&](size_t bytes) -> void* {
        off = (off + 255) & ~(size_t)255;
        void* p = ws + off;
        off += bytes;
        return p;
    };
    auto fbuf = [&](size_t n) -> float*     { return (float*)alloc(n * 4); };
    auto hbuf = [&](size_t n) -> _Float16*  { return (_Float16*)alloc(n * 2); };
    auto ibuf = [&](size_t n) -> int*       { return (int*)alloc(n * 4); };

    float* xbuf  = fbuf((size_t)N*3);
    float* h1t   = fbuf((size_t)N*64);
    float* big0  = fbuf((size_t)N*1024);   // tnet-h2 / fc0out / gat-src
    float* big1  = fbuf((size_t)N*1024);   // gat-dst
    float* gato  = fbuf((size_t)N*1024);
    float* hbig  = fbuf((size_t)N*1408);
    float* tb    = fbuf((size_t)N*64);
    float* pb    = fbuf((size_t)N*64);
    float* h0    = fbuf((size_t)N*64);
    float* h1    = fbuf((size_t)N*64);
    float* h2    = fbuf((size_t)N*64);
    float* hcat  = fbuf((size_t)N*192);
    float* hxy   = fbuf((size_t)N*16);
    float* hyz   = fbuf((size_t)N*16);
    float* hxz   = fbuf((size_t)N*16);
    float* hxyp  = fbuf((size_t)N*64);
    float* hyzp  = fbuf((size_t)N*64);
    float* hxzp  = fbuf((size_t)N*64);
    float* c1    = fbuf((size_t)N*256);
    float* c2    = fbuf((size_t)N*256);
    float* poolt = fbuf(1024);
    float* pool0 = fbuf(1024);
    float* fc3o  = fbuf(256);
    float* m9    = fbuf(9);
    float* qb    = fbuf(9);
    float* rb    = fbuf(9);
    float* cxy   = fbuf((size_t)N*2);
    float* cyz   = fbuf((size_t)N*2);
    float* cxz   = fbuf((size_t)N*2);
    _Float16* a16 = hbuf((size_t)N*1408);
    _Float16* w_fc2t = hbuf(1024*64);
    _Float16* w_g2t = hbuf(64*64); _Float16* w_g2p = hbuf(64*64);
    _Float16* w_g3t = hbuf(64*64); _Float16* w_g3p = hbuf(64*64);
    _Float16* w_g4t = hbuf(64*64); _Float16* w_g4p = hbuf(64*64);
    _Float16* w_g5t = hbuf(64*64); _Float16* w_g5p = hbuf(64*64);
    _Float16* w_ext[3]; _Float16* w_exp[3];
    for (int i = 0; i < 3; ++i) { w_ext[i] = hbuf(64*32); w_exp[i] = hbuf(64*32); }  // K padded 16->32
    _Float16* w_fc0 = hbuf((size_t)1024*192);
    _Float16* w_gws = hbuf((size_t)1024*1408);
    _Float16* w_gwd = hbuf((size_t)1024*1408);
    _Float16* w_fc1 = hbuf((size_t)256*1024);
    _Float16* w_fc2 = hbuf((size_t)256*256);
    _Float16* w_fc3 = hbuf((size_t)64*256);   // rows padded 50->64
    int* idx_xyz = ibuf((size_t)N*30);
    int* idx_xy  = ibuf((size_t)N*30);
    int* idx_yz  = ibuf((size_t)N*30);
    int* idx_xz  = ibuf((size_t)N*30);
    int* g0i     = ibuf((size_t)N*30);
    int* g1i     = ibuf((size_t)N*30);
    int* gxy     = ibuf((size_t)N*100);
    int* gyz     = ibuf((size_t)N*100);
    int* gxz     = ibuf((size_t)N*100);

    auto cvtp = [&](const float* s, _Float16* d, int sr, int sc, int dr, int dc) {
        int tot = dr * dc;
        cvt_pad_kernel<<<(tot + 255) / 256, 256, 0, stream>>>(s, d, sr, sc, dr, dc);
    };
    auto cvt = [&](const float* s, _Float16* d, int rows, int cols) {
        cvtp(s, d, rows, cols, rows, cols);
    };
    // Npad: padded output-column count (W rows, %16==0); Kp: padded K (%32==0)
    auto gemm = [&](const _Float16* A, const _Float16* W, const float* bias,
                    const float* g, const float* b2, float* C,
                    int M, int Npad, int Kp, int Nout, int act) {
        dim3 grid(Npad / 16, M / 16);
        gemm_wmma_kernel<<<grid, 32, 0, stream>>>(A, W, bias, g, b2, C, Kp, Nout, act);
    };
    auto lin = [&](const float* X, const float* W, const float* bias, float* Y,
                   int M, int Nn, int K, int act) {
        int tot = M * Nn;
        linear_small_kernel<<<(tot + 255) / 256, 256, 0, stream>>>(X, W, bias, Y, M, Nn, K, act);
    };
    auto knn = [&](const float* F, int D, int k, int* out) {
        knn_kernel<<<N, 256, 0, stream>>>(F, N, D, k, out);
    };
    auto emax = [&](const float* t, const float* p, const float* bth, const int* id,
                    float* o, int D, int k) {
        edge_max_kernel<<<N, D, 0, stream>>>(t, p, bth, id, o, N, D, k);
    };
    auto ccols = [&](const float* s, int ss, float* d, int ds, int doff, int rows, int cols) {
        int tot = rows * cols;
        copy_cols_kernel<<<(tot + 255) / 256, 256, 0, stream>>>(s, ss, d, ds, doff, rows, cols);
    };

    // ---- weight conversions (cheap, deterministic every call)
    cvt(P(3),  w_fc2t, 1024, 64);
    cvt(P(13), w_g2t, 64, 64); cvt(P(15), w_g2p, 64, 64);
    cvt(P(17), w_g3t, 64, 64); cvt(P(19), w_g3p, 64, 64);
    cvt(P(21), w_g4t, 64, 64); cvt(P(23), w_g4p, 64, 64);
    cvt(P(25), w_g5t, 64, 64); cvt(P(27), w_g5p, 64, 64);
    const int e2base[3] = {33, 41, 49};   // Econv2xy / yz / xz
    for (int i = 0; i < 3; ++i) {
        cvtp(P(e2base[i]),   w_ext[i], 64, 16, 64, 32);   // K zero-padded to 32
        cvtp(P(e2base[i]+2), w_exp[i], 64, 16, 64, 32);
    }
    cvt(P(53), w_fc0, 1024, 192);
    cvt(P(55), w_gws, 1024, 1408);
    cvt(P(56), w_gwd, 1024, 1408);
    cvt(P(59), w_fc1, 256, 1024);
    cvt(P(63), w_fc2, 256, 256);
    cvtp(P(67), w_fc3, 50, 256, 64, 256);                 // rows zero-padded to 64

    // ---- TNet
    lin(P(0), P(1), P(2), h1t, N, 64, 3, 1);                           // fc1 + relu
    cvt(h1t, a16, N, 64);
    gemm(a16, w_fc2t, P(4), nullptr, nullptr, big0, N, 1024, 64, 1024, 0); // fc2
    rowmax_kernel<<<4, 256, 0, stream>>>(big0, poolt, N, 1024);
    lin(poolt, P(5), P(6), fc3o, 1, 256, 1024, 1);                     // fc3 + relu
    lin(fc3o, P(7), P(8), m9, 1, 9, 256, 0);                           // fc4 -> 3x3
    qr3_kernel<<<1, 1, 0, stream>>>(m9, qb, rb);
    rotate_kernel<<<(N + 255) / 256, 256, 0, stream>>>(P(0), qb, xbuf, N);

    // ---- base kNN graphs
    knn(xbuf, 3, 30, idx_xyz);
    extract2_kernel<<<(N+255)/256, 256, 0, stream>>>(xbuf, 0, 1, cxy, N);
    extract2_kernel<<<(N+255)/256, 256, 0, stream>>>(xbuf, 1, 2, cyz, N);
    extract2_kernel<<<(N+255)/256, 256, 0, stream>>>(xbuf, 0, 2, cxz, N);
    knn(cxy, 2, 30, idx_xy);
    knn(cyz, 2, 30, idx_yz);
    knn(cxz, 2, 30, idx_xz);

    // ---- Gconv1 (3->64, f32 path) then Gconv2 (64->64, WMMA)
    lin(xbuf, P(9),  nullptr, tb, N, 64, 3, 0);        // theta (bias folded in emax)
    lin(xbuf, P(11), P(12),   pb, N, 64, 3, 0);        // phi (+bias)
    emax(tb, pb, P(10), idx_xyz, h0, 64, 30);
    cvt(h0, a16, N, 64);
    gemm(a16, w_g2t, nullptr, nullptr, nullptr, tb, N, 64, 64, 64, 0);
    gemm(a16, w_g2p, P(16),   nullptr, nullptr, pb, N, 64, 64, 64, 0);
    emax(tb, pb, P(14), idx_xyz, h0, 64, 30);
    // ---- dynamic graph g0, Gconv3/4
    knn(h0, 64, 30, g0i);
    cvt(h0, a16, N, 64);
    gemm(a16, w_g3t, nullptr, nullptr, nullptr, tb, N, 64, 64, 64, 0);
    gemm(a16, w_g3p, P(20),   nullptr, nullptr, pb, N, 64, 64, 64, 0);
    emax(tb, pb, P(18), g0i, h1, 64, 30);
    cvt(h1, a16, N, 64);
    gemm(a16, w_g4t, nullptr, nullptr, nullptr, tb, N, 64, 64, 64, 0);
    gemm(a16, w_g4p, P(24),   nullptr, nullptr, pb, N, 64, 64, 64, 0);
    emax(tb, pb, P(22), g0i, h1, 64, 30);
    // ---- dynamic graph g1, Gconv5
    knn(h1, 64, 30, g1i);
    cvt(h1, a16, N, 64);
    gemm(a16, w_g5t, nullptr, nullptr, nullptr, tb, N, 64, 64, 64, 0);
    gemm(a16, w_g5p, P(28),   nullptr, nullptr, pb, N, 64, 64, 64, 0);
    emax(tb, pb, P(26), g1i, h2, 64, 30);

    // ---- hcat = [h0|h1|h2] ; fc0 ; pool
    ccols(h0, 64, hcat, 192, 0,   N, 64);
    ccols(h1, 64, hcat, 192, 64,  N, 64);
    ccols(h2, 64, hcat, 192, 128, N, 64);
    cvt(hcat, a16, N, 192);
    gemm(a16, w_fc0, P(54), nullptr, nullptr, big0, N, 1024, 192, 1024, 0);
    rowmax_kernel<<<4, 256, 0, stream>>>(big0, pool0, N, 1024);

    // ---- plane EdgeConv branches (Econv1: 3->16 f32; Econv2: 16->64 WMMA, k=100)
    const int e1base[3] = {29, 37, 45};
    const int* pidx[3]  = {idx_xy, idx_yz, idx_xz};
    int*       pgi[3]   = {gxy, gyz, gxz};
    float*     ph[3]    = {hxy, hyz, hxz};
    float*     php[3]   = {hxyp, hyzp, hxzp};
    for (int br = 0; br < 3; ++br) {
        int b1 = e1base[br], b2 = e2base[br];
        lin(xbuf, P(b1),   nullptr,  tb, N, 16, 3, 0);
        lin(xbuf, P(b1+2), P(b1+3),  pb, N, 16, 3, 0);
        emax(tb, pb, P(b1+1), pidx[br], ph[br], 16, 30);
        knn(ph[br], 16, 100, pgi[br]);
        cvtp(ph[br], a16, N, 16, N, 32);   // K zero-padded to 32
        gemm(a16, w_ext[br], nullptr, nullptr, nullptr, tb, N, 64, 32, 64, 0);
        gemm(a16, w_exp[br], P(b2+3), nullptr, nullptr, pb, N, 64, 32, 64, 0);
        emax(tb, pb, P(b2+1), pgi[br], php[br], 64, 100);
    }

    // ---- hbig = [hcat(192) | broadcast(pool0,1024) | hxy_ | hyz_ | hxz_]  -> [N,1408]
    ccols(hcat, 192, hbig, 1408, 0, N, 192);
    broadcast_row_kernel<<<(N*1024 + 255) / 256, 256, 0, stream>>>(pool0, hbig, 1408, 192, N, 1024);
    ccols(hxyp, 64, hbig, 1408, 1216, N, 64);
    ccols(hyzp, 64, hbig, 1408, 1280, N, 64);
    ccols(hxzp, 64, hbig, 1408, 1344, N, 64);

    // ---- GATv2: src/dst projections (1408->1024 WMMA) + attention
    cvt(hbig, a16, N, 1408);
    gemm(a16, w_gws, nullptr, nullptr, nullptr, big0, N, 1024, 1408, 1024, 0);   // src
    gemm(a16, w_gwd, nullptr, nullptr, nullptr, big1, N, 1024, 1408, 1024, 0);   // dst
    gatv2_kernel<<<dim3(N, 4), 256, 0, stream>>>(big0, big1, idx_xyz, P(57), P(58), gato, N, 30);

    // ---- head MLP: fc1(+bn1,relu) -> fc2(+bn2,relu) -> fc3 -> d_out
    float* out = (float*)d_out;
    cvt(gato, a16, N, 1024);
    gemm(a16, w_fc1, P(60), P(61), P(62), c1, N, 256, 1024, 256, 1);
    cvt(c1, a16, N, 256);
    gemm(a16, w_fc2, P(64), P(65), P(66), c2, N, 256, 256, 256, 1);
    cvt(c2, a16, N, 256);
    gemm(a16, w_fc3, P(68), nullptr, nullptr, out, N, 64, 256, 50, 0);

    // ---- append r (3x3) after h [N,50]
    write_r_kernel<<<1, 32, 0, stream>>>(rb, out + (size_t)N * 50);
}
